// VectorQuantizer_73967926772541
// MI455X (gfx1250) — matrix operand order repaired and checked by reading, not compiled, not run on hardware
//
#include <hip/hip_runtime.h>
#include <hip/hip_bf16.h>
#include <cmath>

typedef __attribute__((ext_vector_type(16))) _Float16 v16h;
typedef __attribute__((ext_vector_type(8)))  float    v8f;

#define N_ROWS 65536
#define DIM    513
#define KD     512
#define CODES  1024
#define RBINS  16
#define NE     16384
#define EPSV   1e-7f
#define MAXR   18.0f
#define BETA_C 0.25f

// workspace layout (float offsets)
#define WS_COUNTS 0                       // 16384 floats
#define WS_LOSS   16384
#define WS_H      16385
#define WS_USE    16386
#define WS_INVN   16400                   // 65536 floats
#define WS_SCALE  (16400 + 65536)         // 65536 floats
#define WS_RIDX   (16400 + 2*65536)       // 65536 ints
#define WS_WIDX   (16400 + 3*65536)       // 65536 ints
#define WS_AWF16  (16400 + 4*65536)       // 1024*512 _Float16 (32B aligned)

// ---- CDNA5 async copy helpers (ASYNCcnt path) -------------------------------
__device__ __forceinline__ void async_g2lds_b128(unsigned lds_byte_addr,
                                                 const void* gaddr) {
    asm volatile("global_load_async_to_lds_b128 %0, %1, off"
                 :
                 : "v"(lds_byte_addr),
                   "v"((unsigned long long)(size_t)gaddr)
                 : "memory");
}
__device__ __forceinline__ void wait_asynccnt0() {
#if __has_builtin(__builtin_amdgcn_s_wait_asynccnt)
    __builtin_amdgcn_s_wait_asynccnt(0);
#else
    asm volatile("s_wait_asynccnt 0" ::: "memory");
#endif
}
// -----------------------------------------------------------------------------

__global__ void k_zero(float* ws) {
    int i = blockIdx.x * blockDim.x + threadIdx.x;
    if (i < NE) ws[WS_COUNTS + i] = 0.0f;
    if (i < 3)  ws[WS_LOSS + i]   = 0.0f;
}

// one wave32 per row: inverse norm of u_space, radius, radial-bin argmin, scale
__global__ void k_prep(const float* __restrict__ u, const float* __restrict__ rc_in,
                       float* ws) {
    int wave = (blockIdx.x * blockDim.x + threadIdx.x) >> 5;
    int lane = threadIdx.x & 31;
    if (wave >= N_ROWS) return;
    const float* up = u + (size_t)wave * DIM + 1;
    float ss = 0.0f;
#pragma unroll
    for (int t = 0; t < 16; ++t) { float v = up[lane + 32 * t]; ss += v * v; }
    for (int off = 16; off; off >>= 1) ss += __shfl_xor(ss, off, 32);
    float inv = 1.0f / fmaxf(sqrtf(ss), 1e-12f);
    if (lane == 0) {
        float u0 = fmaxf(u[(size_t)wave * DIM], 1.0f + EPSV);
        float r  = logf(u0 + sqrtf(u0 * u0 - 1.0f));   // arccosh
        int best = 0; float bd = 3.4e38f, brc = 0.01f;
        for (int j = 0; j < RBINS; ++j) {
            float rcv = fminf(fmaxf(rc_in[j], 0.01f), MAXR);
            float d = (r - rcv) * (r - rcv);
            if (d < bd) { bd = d; best = j; brc = rcv; }
        }
        ws[WS_INVN + wave]          = inv;
        ws[WS_SCALE + wave]         = tanhf(brc * 0.5f);
        ((int*)ws)[WS_RIDX + wave]  = best;
    }
}

// pre-swizzle codebook into WMMA B-fragment order:
// linear idx = ((nt*16+kt)*32 + lane)*16 + h ; B 32x16 f16 layout:
// lane -> column n = nt*16 + (lane&15); k = kt*32 + 16*(h>=8) + 8*(lane>=16) + (h&7)
// => each N-tile nt occupies one contiguous 16KB slab (perfect async-copy unit)
__global__ void k_swz(const float* __restrict__ aw, float* ws) {
    int idx = blockIdx.x * blockDim.x + threadIdx.x;
    if (idx >= CODES * KD) return;
    int h  = idx & 15;
    int L  = (idx >> 4) & 31;
    int kt = (idx >> 9) & 15;
    int nt = idx >> 13;
    int v = h >> 1, p = h & 1;
    int n = nt * 16 + (L & 15);
    int k = kt * 32 + ((h & 8) ? 16 : 0) + ((L & 16) ? 8 : 0) + (h & 7); (void)v; (void)p;
    _Float16* dst = (_Float16*)(ws + WS_AWF16);
    dst[idx] = (_Float16)aw[n * KD + k];
}

// 8 waves / block, one 16-row M-tile per wave (128 rows / block).
// B slabs (16KB per N-tile) double-buffered in LDS via async DMA, shared by
// all 8 waves; inner loop = 16x ds_load_b128-pair + v_wmma per slab.
__global__ void __launch_bounds__(256) k_gemm(const float* __restrict__ u, float* ws) {
    __shared__ __align__(32) char smem[2][16384];

    int lane    = threadIdx.x & 31;
    int wave    = threadIdx.x >> 5;
    int rowBase = (blockIdx.x * 8 + wave) * 16;
    int row     = rowBase + (lane & 15);
    const float* up = u + (size_t)row * DIM + 1;
    float inv = ws[WS_INVN + row];

    // A fragments: 16x32 f16 layout (lanes 16-31 hold K+8; VGPR4-7 hold K+16)
    v16h afrag[16];
#pragma unroll
    for (int kt = 0; kt < 16; ++kt) {
#pragma unroll
        for (int v = 0; v < 8; ++v) {
            int k = kt * 32 + ((v & 4) ? 16 : 0) + ((lane & 16) ? 8 : 0) + ((v & 3) << 1);
            afrag[kt][2 * v]     = (_Float16)(up[k]     * inv);
            afrag[kt][2 * v + 1] = (_Float16)(up[k + 1] * inv);
        }
    }

    const char* gB = (const char*)(ws + WS_AWF16);

    // prefetch slab 0 into buffer 0 (256 threads x 4 x 16B = 16KB)
    {
        const char* src = gB + threadIdx.x * 16;
        unsigned    dst = (unsigned)(size_t)&smem[0][threadIdx.x * 16];
#pragma unroll
        for (int i = 0; i < 4; ++i)
            async_g2lds_b128(dst + i * 4096, src + i * 4096);
    }
    wait_asynccnt0();
    __syncthreads();

    float bestv[8]; int bestn[8];
#pragma unroll
    for (int r = 0; r < 8; ++r) { bestv[r] = -3.4e38f; bestn[r] = 0; }

    for (int nt = 0; nt < 64; ++nt) {
        // kick off DMA for the next slab while computing this one
        if (nt < 63) {
            const char* src = gB + (size_t)(nt + 1) * 16384 + threadIdx.x * 16;
            unsigned    dst = (unsigned)(size_t)&smem[(nt + 1) & 1][threadIdx.x * 16];
#pragma unroll
            for (int i = 0; i < 4; ++i)
                async_g2lds_b128(dst + i * 4096, src + i * 4096);
        }

        const char* bbase = &smem[nt & 1][lane * 32];
        v8f acc = {0.f, 0.f, 0.f, 0.f, 0.f, 0.f, 0.f, 0.f};
#pragma unroll
        for (int kt = 0; kt < 16; ++kt) {
            v16h b = *(const v16h*)(bbase + kt * 1024);
            acc = __builtin_amdgcn_wmma_f32_16x16x32_f16(
                false, afrag[kt], false, b, (short)0, acc, false, false);
        }

        int n = nt * 16 + (lane & 15);       // C/D layout: lane -> N
#pragma unroll
        for (int r = 0; r < 8; ++r)          // VGPR index -> M (+8 for upper half)
            if (acc[r] > bestv[r]) { bestv[r] = acc[r]; bestn[r] = n; }

        if (nt < 63) { wait_asynccnt0(); __syncthreads(); }
    }

    // reduce argmax across the 16 lanes of each half (tie -> lower index)
#pragma unroll
    for (int off = 1; off <= 8; off <<= 1) {
#pragma unroll
        for (int r = 0; r < 8; ++r) {
            float ov = __shfl_xor(bestv[r], off, 32);
            int   on = __shfl_xor(bestn[r], off, 32);
            if (ov > bestv[r] || (ov == bestv[r] && on < bestn[r])) {
                bestv[r] = ov; bestn[r] = on;
            }
        }
    }
    if ((lane & 15) == 0) {
        int half = (lane >> 4) & 1;
#pragma unroll
        for (int r = 0; r < 8; ++r)
            ((int*)ws)[WS_WIDX + rowBase + r + 8 * half] = bestn[r];
    }
}

// one wave per row: rebuild x_q in f32, write z_q, loss + histogram atomics
__global__ void k_fin(const float* __restrict__ u, const float* __restrict__ aw,
                      float* ws, float* out) {
    int wave = (blockIdx.x * blockDim.x + threadIdx.x) >> 5;
    int lane = threadIdx.x & 31;
    if (wave >= N_ROWS) return;
    int row = wave;
    float s = ws[WS_SCALE + row];
    int   wi = ((int*)ws)[WS_WIDX + row];
    int   ri = ((int*)ws)[WS_RIDX + row];
    const float* ap = aw + (size_t)wi * KD;
    float xp[16]; float sq = 0.0f;
#pragma unroll
    for (int t = 0; t < 16; ++t) { xp[t] = s * ap[lane + 32 * t]; sq += xp[t] * xp[t]; }
    for (int off = 16; off; off >>= 1) sq += __shfl_xor(sq, off, 32);
    float denom = fmaxf(1.0f - sq, EPSV);
    float c = 2.0f / denom;
    const float* up = u + (size_t)row * DIM;
    float* zrow = out + 1 + (size_t)row * DIM;
    float t2 = 0.0f, du = 0.0f;
#pragma unroll
    for (int t = 0; t < 16; ++t) {
        float spc = c * xp[t];
        t2 += spc * spc;
        du += up[1 + lane + 32 * t] * spc;
        zrow[1 + lane + 32 * t] = spc;
    }
    for (int off = 16; off; off >>= 1) {
        t2 += __shfl_xor(t2, off, 32);
        du += __shfl_xor(du, off, 32);
    }
    float tm = sqrtf(1.0f + t2);
    if (lane == 0) {
        zrow[0] = tm;
        float inner = fmaxf(up[0] * tm - du, 1.0f + EPSV);
        float dist  = logf(inner + sqrtf(inner * inner - 1.0f));
        atomicAdd(ws + WS_LOSS, dist);
        atomicAdd(ws + WS_COUNTS + ri * CODES + wi, 1.0f);
    }
}

__global__ void k_stats(float* ws, float* out) {
    int i = blockIdx.x * blockDim.x + threadIdx.x;
    if (i >= NE) return;
    float em = ws[WS_COUNTS + i] * (1.0f / (float)N_ROWS);
    out[1 + (size_t)N_ROWS * DIM + 2 + i] = em;          // e_mean
    atomicAdd(ws + WS_H, em * logf(em + 1e-10f));
    if (em > 0.0f) atomicAdd(ws + WS_USE, 1.0f);
}

__global__ void k_final_scalars(float* ws, float* out) {
    out[0] = BETA_C * ws[WS_LOSS] * (1.0f / (float)N_ROWS);      // loss
    size_t p = 1 + (size_t)N_ROWS * DIM;
    out[p]     = expf(-ws[WS_H]);                                 // perplexity
    out[p + 1] = ws[WS_USE] * (1.0f / (float)NE);                 // usage
}

extern "C" void kernel_launch(void* const* d_in, const int* in_sizes, int n_in,
                              void* d_out, int out_size, void* d_ws, size_t ws_size,
                              hipStream_t stream) {
    const float* u  = (const float*)d_in[0];   // (64,32,32,513) f32
    const float* rc = (const float*)d_in[1];   // (16,) f32
    const float* aw = (const float*)d_in[2];   // (1024,512) f32
    float* out = (float*)d_out;
    float* ws  = (float*)d_ws;
    (void)in_sizes; (void)n_in; (void)out_size; (void)ws_size;

    k_zero<<<(NE + 255) / 256, 256, 0, stream>>>(ws);
    k_prep<<<(N_ROWS * 32) / 256, 256, 0, stream>>>(u, rc, ws);
    k_swz<<<(CODES * KD) / 256, 256, 0, stream>>>(aw, ws);
    k_gemm<<<N_ROWS / 128, 256, 0, stream>>>(u, ws);
    k_fin<<<(N_ROWS * 32) / 256, 256, 0, stream>>>(u, aw, ws, out);
    k_stats<<<NE / 256, 256, 0, stream>>>(ws, out);
    k_final_scalars<<<1, 1, 0, stream>>>(ws, out);
}
